// CGAT_86285892976695
// MI455X (gfx1250) — matrix operand order repaired and checked by reading, not compiled, not run on hardware
//
#include <hip/hip_runtime.h>

// ---------------------------------------------------------------------------
// CGAT forward for MI455X (gfx1250, wave32, WMMA).
// Pipeline:
//   conv1 (VALU, recompute for BN stats) -> pool1 bf16 NHWC
//   conv2 (implicit-GEMM, v_wmma_f32_16x16x32_bf16) -> BN stats -> pool -> x0
//   3x GAT layer: WMMA GEMM (features + attention cols fused) ->
//                 global max (ordered-uint atomicMax) -> exp+scatter atomics ->
//                 normalize (-> bf16 next-layer features / graph segment-sum)
//   tiny MLP head -> d_out [32,10] fp32
// ---------------------------------------------------------------------------

typedef unsigned int  u32;
typedef unsigned short u16;
typedef __attribute__((ext_vector_type(16))) __bf16 v16bf;
typedef __attribute__((ext_vector_type(8)))  float  v8f;
typedef __attribute__((ext_vector_type(4)))  u32    u32x4;

union Frag16 { v16bf v; u32x4 q[2]; u32 u[8]; };

__device__ __forceinline__ u16 f2bf(float f) {
  u32 u = __float_as_uint(f);
  u32 r = u + 0x7FFFu + ((u >> 16) & 1u);   // round-to-nearest-even
  return (u16)(r >> 16);
}
__device__ __forceinline__ float bf2f(u16 h) { return __uint_as_float(((u32)h) << 16); }
__device__ __forceinline__ u32 fkey(float f) {
  u32 u = __float_as_uint(f);
  return (u & 0x80000000u) ? ~u : (u | 0x80000000u);
}
__device__ __forceinline__ float fkeyinv(u32 k) {
  u32 u = (k & 0x80000000u) ? (k & 0x7FFFFFFFu) : ~k;
  return __uint_as_float(u);
}
__device__ __forceinline__ v8f vzero8() {
  v8f z = {0.f, 0.f, 0.f, 0.f, 0.f, 0.f, 0.f, 0.f};
  return z;
}

// ---------------- problem constants ----------------
#define BN_   32          // batch
#define HIN_  262
#define H1_   260         // conv1 output side
#define P1_   130         // pool1 side
#define H2_   128         // conv2 output side
#define NND_  131072      // nodes  (32*64*64)
#define EED_  393216      // edges  (3*NND)
#define M1F_  2163200.0f  // BN1 count = 32*260*260
#define M2F_  524288.0f   // BN2 count = 32*128*128

// ---------------- workspace layout (bytes) ----------------
// num fp32 [N,128] = 64 MiB; pool1 bf16 (34.6 MB) overlays (dead before num used)
// y bf16 [E,128] = 96 MiB; conv2 raw fp32 (64 MiB) overlays (dead before y used)
static const size_t OFF_NUM  = 0;
static const size_t OFF_Y    = 67108864;
static const size_t OFF_XA   = OFF_Y   + 100663296;
static const size_t OFF_XB   = OFF_XA  + 33554432;
static const size_t OFF_A    = OFF_XB  + 33554432;
static const size_t OFF_DEN  = OFF_A   + 3145728;
static const size_t OFF_MISC = OFF_DEN + 1048576;

// ------------------------------------------------------------------
__global__ void zerof(float* __restrict__ p, int n) {
  int stride = gridDim.x * blockDim.x;
  for (int i = blockIdx.x * blockDim.x + threadIdx.x; i < n; i += stride) p[i] = 0.f;
}

__global__ void init_amax(u32* a) {
  if (threadIdx.x < 2) a[threadIdx.x] = 0x007FFFFFu;  // fkey(-inf)
}

// ------------------------------------------------------------------
// conv1 statistics: recompute conv+bias, accumulate per-channel sum/sumsq
__global__ __launch_bounds__(256)
void conv1_stats(const float* __restrict__ x, const float* __restrict__ W1,
                 const float* __restrict__ b1, float* __restrict__ stats) {
  __shared__ float wls[864];
  __shared__ float b1s[32];
  __shared__ float bs[64];
  for (int i = threadIdx.x; i < 864; i += 256) wls[i] = W1[i];
  if (threadIdx.x < 32) b1s[threadIdx.x] = b1[threadIdx.x];
  if (threadIdx.x < 64) bs[threadIdx.x] = 0.f;
  __syncthreads();
  int p = blockIdx.x * 256 + threadIdx.x;     // grid exactly covers 32*260*260
  int b = p / (H1_ * H1_);
  int rem = p - b * (H1_ * H1_);
  int oy = rem / H1_, ox = rem % H1_;
  float pin[3][3][3];
  for (int c = 0; c < 3; ++c)
    for (int ky = 0; ky < 3; ++ky)
      for (int kx = 0; kx < 3; ++kx)
        pin[c][ky][kx] = x[((size_t)(b * 3 + c) * HIN_ + oy + ky) * HIN_ + ox + kx];
  float sum[32], sq[32];
  for (int co = 0; co < 32; ++co) {
    float acc = b1s[co];
    const float* wc = &wls[co * 27];
#pragma unroll
    for (int c = 0; c < 3; ++c)
#pragma unroll
      for (int ky = 0; ky < 3; ++ky)
#pragma unroll
        for (int kx = 0; kx < 3; ++kx)
          acc = fmaf(pin[c][ky][kx], wc[(c * 3 + ky) * 3 + kx], acc);
    sum[co] = acc;
    sq[co]  = acc * acc;
  }
  for (int co = 0; co < 32; ++co) {
    atomicAdd(&bs[co], sum[co]);
    atomicAdd(&bs[32 + co], sq[co]);
  }
  __syncthreads();
  if (threadIdx.x < 64) atomicAdd(&stats[threadIdx.x], bs[threadIdx.x]);
}

// conv1 + BN + ReLU + 2x2 maxpool, output NHWC bf16 [32,130,130,32]
__global__ __launch_bounds__(256)
void conv1_bnpool(const float* __restrict__ x, const float* __restrict__ W1,
                  const float* __restrict__ b1, const float* __restrict__ g1,
                  const float* __restrict__ be1, const float* __restrict__ stats,
                  u16* __restrict__ pool1) {
  __shared__ float wls[864], b1s[32], scl[32], shf[32];
  for (int i = threadIdx.x; i < 864; i += 256) wls[i] = W1[i];
  if (threadIdx.x < 32) {
    int c = threadIdx.x;
    b1s[c] = b1[c];
    float mu  = stats[c] / M1F_;
    float var = stats[32 + c] / M1F_ - mu * mu;
    float sc  = g1[c] * rsqrtf(var + 1e-5f);
    scl[c] = sc;
    shf[c] = be1[c] - mu * sc;
  }
  __syncthreads();
  int p = blockIdx.x * 256 + threadIdx.x;
  if (p >= BN_ * P1_ * P1_) return;
  int b = p / (P1_ * P1_);
  int rem = p - b * (P1_ * P1_);
  int py = rem / P1_, px = rem % P1_;
  float pin[3][4][4];
  for (int c = 0; c < 3; ++c)
    for (int iy = 0; iy < 4; ++iy)
      for (int ix = 0; ix < 4; ++ix)
        pin[c][iy][ix] = x[((size_t)(b * 3 + c) * HIN_ + 2 * py + iy) * HIN_ + 2 * px + ix];
  for (int co = 0; co < 32; ++co) {
    const float* wc = &wls[co * 27];
    float mx = 0.f;  // post-ReLU values are >= 0
#pragma unroll
    for (int dy = 0; dy < 2; ++dy)
#pragma unroll
      for (int dx = 0; dx < 2; ++dx) {
        float acc = b1s[co];
#pragma unroll
        for (int c = 0; c < 3; ++c)
#pragma unroll
          for (int ky = 0; ky < 3; ++ky)
#pragma unroll
            for (int kx = 0; kx < 3; ++kx)
              acc = fmaf(pin[c][dy + ky][dx + kx], wc[(c * 3 + ky) * 3 + kx], acc);
        float v = acc * scl[co] + shf[co];
        v = v > 0.f ? v : 0.f;
        mx = fmaxf(mx, v);
      }
    pool1[(size_t)p * 32 + co] = f2bf(mx);
  }
}

// ------------------------------------------------------------------
// conv2 weight repack: Wc2[k=(kh*3+kw)*32+ci][co] bf16
__global__ void prep_w2(const float* __restrict__ W2, u16* __restrict__ Wc2) {
  int t = blockIdx.x * 256 + threadIdx.x;
  if (t >= 288 * 32) return;
  int k = t >> 5, co = t & 31;
  int khw = k >> 5, ci = k & 31;
  int kh = khw / 3, kw = khw % 3;
  Wc2[(size_t)k * 32 + co] = f2bf(W2[((size_t)(co * 32 + ci) * 3 + kh) * 3 + kw]);
}

// conv2 implicit GEMM: M=524288 pixels, K=288, N=32; bf16 WMMA, f32 accum
__global__ __launch_bounds__(256)
void conv2_gemm(const u16* __restrict__ pool1, const u16* __restrict__ Wc2,
                const float* __restrict__ b2, float* __restrict__ c2) {
  extern __shared__ unsigned char smem_raw[];
  u16* w = (u16*)smem_raw;                 // [288][34] bf16, odd-dword stride
  const u32* gw = (const u32*)Wc2;
  for (int i = threadIdx.x; i < 288 * 16; i += 256) {
    int row = i >> 4, c = i & 15;
    ((u32*)w)[row * 17 + c] = gw[i];
  }
  __syncthreads();
  int warp = threadIdx.x >> 5, lane = threadIdx.x & 31;
  int hi = lane >> 4, m = lane & 15;
  int base = (blockIdx.x * 8 + warp) * 16;
  int p = base + m;
  int b = p >> 14, rem = p & 16383;
  int yy = rem >> 7, xx = rem & 127;
  const u16* inb = pool1 + ((size_t)(b * P1_ + yy) * P1_ + xx) * 32;
  v8f acc[2];
  acc[0] = vzero8(); acc[1] = vzero8();
#pragma unroll
  for (int kb = 0; kb < 9; ++kb) {
    int kh = kb / 3, kw = kb - kh * 3;
    const u16* pr = inb + ((size_t)kh * P1_ + kw) * 32;
    Frag16 af;
    af.q[0] = *(const u32x4*)(pr + hi * 8);
    af.q[1] = *(const u32x4*)(pr + 16 + hi * 8);
    int krow = kb * 32 + lane;
    const u32* wrow = ((const u32*)w) + krow * 17;
#pragma unroll
    for (int nb = 0; nb < 2; ++nb) {
      Frag16 bf;
#pragma unroll
      for (int j = 0; j < 8; ++j) { int kk = kb * 32 + hi * 8 + 2 * j + (j >= 4 ? 8 : 0); bf.u[j] = (u32)w[kk * 34 + nb * 16 + m] | ((u32)w[(kk + 1) * 34 + nb * 16 + m] << 16); }
      acc[nb] = __builtin_amdgcn_wmma_f32_16x16x32_bf16(false, af.v, false, bf.v,
                                                        (short)0, acc[nb], false, false);
    }
  }
#pragma unroll
  for (int nb = 0; nb < 2; ++nb) {
    int col = nb * 16 + (lane & 15);
    float bias = b2[col];
#pragma unroll
    for (int r = 0; r < 8; ++r) {
      int pe = base + r + hi * 8;
      c2[(size_t)pe * 32 + col] = acc[nb][r] + bias;
    }
  }
}

__global__ void conv2_stats(const float* __restrict__ c2, float* __restrict__ stats) {
  int tid = blockIdx.x * 256 + threadIdx.x;   // 524288 threads
  int ch = tid & 31;
  float s = 0.f, q = 0.f;
  for (int r = tid >> 5; r < BN_ * H2_ * H2_; r += 16384) {
    float v = c2[(size_t)r * 32 + ch];
    s += v;
    q = fmaf(v, v, q);
  }
  atomicAdd(&stats[ch], s);
  atomicAdd(&stats[32 + ch], q);
}

// BN + ReLU + pool -> node features x0 bf16 [N,32]
__global__ __launch_bounds__(256)
void bnpool2(const float* __restrict__ c2, const float* __restrict__ g2,
             const float* __restrict__ be2, const float* __restrict__ stats,
             u16* __restrict__ x0) {
  __shared__ float scl[32], shf[32];
  if (threadIdx.x < 32) {
    int c = threadIdx.x;
    float mu  = stats[c] / M2F_;
    float var = stats[32 + c] / M2F_ - mu * mu;
    float sc  = g2[c] * rsqrtf(var + 1e-5f);
    scl[c] = sc;
    shf[c] = be2[c] - mu * sc;
  }
  __syncthreads();
  int n = blockIdx.x * 256 + threadIdx.x;  // exact
  int b = n >> 12, rem = n & 4095;
  int py = rem >> 6, px = rem & 63;
  const float* base = c2 + ((size_t)(b * H2_ + 2 * py) * H2_ + 2 * px) * 32;
  for (int co = 0; co < 32; ++co) {
    float sc = scl[co], sh = shf[co];
    float v0 = base[co] * sc + sh;
    float v1 = base[32 + co] * sc + sh;
    float v2 = base[H2_ * 32 + co] * sc + sh;
    float v3 = base[H2_ * 32 + 32 + co] * sc + sh;
    v0 = v0 > 0.f ? v0 : 0.f;  v1 = v1 > 0.f ? v1 : 0.f;
    v2 = v2 > 0.f ? v2 : 0.f;  v3 = v3 > 0.f ? v3 : 0.f;
    x0[(size_t)n * 32 + co] = f2bf(fmaxf(fmaxf(v0, v1), fmaxf(v2, v3)));
  }
}

// ------------------------------------------------------------------
// GAT weight pack: Wcat[k][c] bf16, cols [0,HF)=fW^T per head, HF/HF+1 = wW, rest 0
__global__ void prep_wcat(const float* __restrict__ fW, const float* __restrict__ fb,
                          const float* __restrict__ wW, const float* __restrict__ wb,
                          u16* __restrict__ Wcat, float* __restrict__ biascat,
                          int K2, int F, int HF, int NC) {
  int t = blockIdx.x * 256 + threadIdx.x;
  if (t >= K2 * NC) return;
  int k = t / NC, c = t - k * NC;
  float v = 0.f;
  if (c < HF)            { int h = c / F, j = c - h * F; v = fW[((size_t)(h * F + j)) * K2 + k]; }
  else if (c < HF + 2)   { v = wW[(size_t)(c - HF) * K2 + k]; }
  Wcat[(size_t)k * NC + c] = f2bf(v);
  if (k == 0) {
    float bv = 0.f;
    if (c < HF)          { int h = c / F, j = c - h * F; bv = fb[h * F + j]; }
    else if (c < HF + 2) { bv = wb[c - HF]; }
    biascat[c] = bv;
  }
}

// GAT GEMM: per wave one 16-edge tile; A gathered from bf16 node features
// (concat src/tgt rows per documented 16-bit A layout), B (weights+attention
// columns) staged in LDS, f32 WMMA accumulators, fused ReLU/bias/amax.
template <int KS, int NT, int HF, int D>
__global__ __launch_bounds__(256)
void gat_gemm(const u16* __restrict__ xin, const int* __restrict__ src,
              const int* __restrict__ tgt, const u16* __restrict__ Wcat,
              const float* __restrict__ biascat, u16* __restrict__ y,
              float* __restrict__ a_out, u32* __restrict__ amaxU) {
  const int NC = NT * 16;
  const int SU = (NC + 2) / 2;  // odd dword stride -> conflict-free B reads
  extern __shared__ unsigned char smem_raw[];
  u16* w = (u16*)smem_raw;      // [2D][NC+2]
  const u32* gw = (const u32*)Wcat;
  for (int i = threadIdx.x; i < KS * 32 * (NC / 2); i += 256) {
    int row = i / (NC / 2), c = i - row * (NC / 2);
    ((u32*)w)[row * SU + c] = gw[i];
  }
  __syncthreads();
  int warp = threadIdx.x >> 5, lane = threadIdx.x & 31;
  int hi = lane >> 4, m = lane & 15;
  int et = (blockIdx.x * 8 + warp) * 16;
  int e = et + m;
  const u16* xs = xin + (size_t)src[e] * D;
  const u16* xt = xin + (size_t)tgt[e] * D;
  v8f acc[NT];
#pragma unroll
  for (int n = 0; n < NT; ++n) acc[n] = vzero8();
#pragma unroll
  for (int kb = 0; kb < KS; ++kb) {
    Frag16 af;
    int k1 = kb * 32 + hi * 8;
    af.q[0] = *(const u32x4*)((k1 < D) ? (xs + k1) : (xt + (k1 - D)));
    int k2 = k1 + 16;
    af.q[1] = *(const u32x4*)((k2 < D) ? (xs + k2) : (xt + (k2 - D)));
    int krow = kb * 32 + lane;
    const u32* wrow = ((const u32*)w) + krow * SU;
#pragma unroll
    for (int nb = 0; nb < NT; ++nb) {
      Frag16 bf;
#pragma unroll
      for (int j = 0; j < 8; ++j) { int kk = kb * 32 + hi * 8 + 2 * j + (j >= 4 ? 8 : 0); bf.u[j] = (u32)w[kk * (2 * SU) + nb * 16 + m] | ((u32)w[(kk + 1) * (2 * SU) + nb * 16 + m] << 16); }
      acc[nb] = __builtin_amdgcn_wmma_f32_16x16x32_bf16(false, af.v, false, bf.v,
                                                        (short)0, acc[nb], false, false);
    }
  }
  int colb = lane & 15;
#pragma unroll
  for (int nb = 0; nb < NT - 1; ++nb) {  // feature columns: bias + ReLU -> bf16
    int col = nb * 16 + colb;
    float bias = biascat[col];
#pragma unroll
    for (int r = 0; r < 8; ++r) {
      int er = et + r + hi * 8;
      float v = acc[nb][r] + bias;
      v = v > 0.f ? v : 0.f;
      y[(size_t)er * HF + col] = f2bf(v);
    }
  }
  if (colb < 2) {                        // attention columns (tile NT-1, cols 0/1)
    float bias = biascat[HF + colb];
    float lm = -3.4e38f;
#pragma unroll
    for (int r = 0; r < 8; ++r) {
      int er = et + r + hi * 8;
      float v = acc[NT - 1][r] + bias;
      a_out[(size_t)er * 2 + colb] = v;
      lm = fmaxf(lm, v);
    }
    atomicMax(&amaxU[colb], fkey(lm));
  }
}

// edge softmax scatter: w=exp(a-amax); den[tgt]+=w; num[tgt]+=y*w
__global__ void gat_scatter(const u16* __restrict__ y, const float* __restrict__ a,
                            const u32* __restrict__ amaxU, const int* __restrict__ tgt,
                            float* __restrict__ num, float* __restrict__ den,
                            int F, int HF) {
  int t = blockIdx.x * 256 + threadIdx.x;  // E*2 exact
  int e = t >> 1, h = t & 1;
  float am = fkeyinv(amaxU[h]);
  float wv = expf(a[(size_t)e * 2 + h] - am);
  int tg = tgt[e];
  atomicAdd(&den[(size_t)tg * 2 + h], wv);
  const u16* yr = y + (size_t)e * HF + h * F;
  float* nr = num + (size_t)tg * HF + h * F;
  for (int j = 0; j < F; ++j) atomicAdd(&nr[j], bf2f(yr[j]) * wv);
}

// normalize: x = num/(den+eps); emit bf16 features, or graph segment-sum (L3)
__global__ void gat_norm(const float* __restrict__ num, const float* __restrict__ den,
                         u16* __restrict__ xout, float* __restrict__ gsum,
                         int F, int HF) {
  int idx = blockIdx.x * 256 + threadIdx.x;  // N*HF exact
  int n = idx / HF, c = idx - n * HF;
  int h = c / F;
  float v = num[idx] / (den[(size_t)n * 2 + h] + 1e-6f);
  if (xout) xout[idx] = f2bf(v);
  else      atomicAdd(&gsum[(n >> 12) * HF + c], v);
}

// ------------------------------------------------------------------
__global__ __launch_bounds__(256)
void mlp_head(const float* __restrict__ gsum, const float* __restrict__ m1W,
              const float* __restrict__ m1b, const float* __restrict__ m2W,
              const float* __restrict__ m2b, float* __restrict__ out) {
  __shared__ float sg[32 * 128];
  __shared__ float hid[32 * 32];
  for (int i = threadIdx.x; i < 32 * 128; i += 256) sg[i] = gsum[i];
  __syncthreads();
  for (int o = threadIdx.x; o < 32 * 32; o += 256) {
    int i = o >> 5, j = o & 31;
    float acc = m1b[j];
    for (int c = 0; c < 128; ++c) acc = fmaf(sg[i * 128 + c], m1W[j * 128 + c], acc);
    hid[o] = acc > 0.f ? acc : 0.f;
  }
  __syncthreads();
  for (int o = threadIdx.x; o < 320; o += 256) {
    int i = o / 10, k = o - (o / 10) * 10;
    float acc = m2b[k];
    for (int j = 0; j < 32; ++j) acc = fmaf(hid[i * 32 + j], m2W[k * 32 + j], acc);
    out[i * 10 + k] = acc;
  }
}

// ------------------------------------------------------------------
extern "C" void kernel_launch(void* const* d_in, const int* in_sizes, int n_in,
                              void* d_out, int out_size, void* d_ws, size_t ws_size,
                              hipStream_t stream) {
  (void)in_sizes; (void)n_in; (void)out_size; (void)ws_size;
  const float* x   = (const float*)d_in[0];
  const float* W1  = (const float*)d_in[1];
  const float* b1  = (const float*)d_in[2];
  const float* g1  = (const float*)d_in[3];
  const float* be1 = (const float*)d_in[4];
  const float* W2  = (const float*)d_in[5];
  const float* b2  = (const float*)d_in[6];
  const float* g2  = (const float*)d_in[7];
  const float* be2 = (const float*)d_in[8];
  const float* fW[3] = {(const float*)d_in[9],  (const float*)d_in[13], (const float*)d_in[17]};
  const float* fb[3] = {(const float*)d_in[10], (const float*)d_in[14], (const float*)d_in[18]};
  const float* wW[3] = {(const float*)d_in[11], (const float*)d_in[15], (const float*)d_in[19]};
  const float* wb[3] = {(const float*)d_in[12], (const float*)d_in[16], (const float*)d_in[20]};
  const float* m1W = (const float*)d_in[21];
  const float* m1b = (const float*)d_in[22];
  const float* m2W = (const float*)d_in[23];
  const float* m2b = (const float*)d_in[24];
  const int* src = (const int*)d_in[25];
  const int* tgt = (const int*)d_in[26];
  float* out = (float*)d_out;

  char* ws = (char*)d_ws;
  float* num    = (float*)(ws + OFF_NUM);
  u16*   pool1  = (u16*)  (ws + OFF_NUM);   // overlay, dead before num is used
  u16*   ybuf   = (u16*)  (ws + OFF_Y);
  float* c2raw  = (float*)(ws + OFF_Y);     // overlay, dead before y is used
  u16*   xA     = (u16*)  (ws + OFF_XA);
  u16*   xB     = (u16*)  (ws + OFF_XB);
  float* abuf   = (float*)(ws + OFF_A);
  float* den    = (float*)(ws + OFF_DEN);
  float* stats1 = (float*)(ws + OFF_MISC);
  float* stats2 = (float*)(ws + OFF_MISC + 256);
  u32*   amaxU  = (u32*)  (ws + OFF_MISC + 512);
  float* biascat= (float*)(ws + OFF_MISC + 768);
  float* gsum   = (float*)(ws + OFF_MISC + 2048);
  u16*   Wcat   = (u16*)  (ws + OFF_MISC + 2048 + 16384);
  u16*   Wc2    = (u16*)  (ws + OFF_MISC + 2048 + 16384 + 73728);

  // ---- conv front-end ----
  zerof<<<64, 256, 0, stream>>>(stats1, 128);  // stats1 + stats2 (contiguous)
  conv1_stats<<<(BN_ * H1_ * H1_) / 256, 256, 0, stream>>>(x, W1, b1, stats1);
  conv1_bnpool<<<(BN_ * P1_ * P1_ + 255) / 256, 256, 0, stream>>>(x, W1, b1, g1, be1,
                                                                  stats1, pool1);
  prep_w2<<<(288 * 32 + 255) / 256, 256, 0, stream>>>(W2, Wc2);
  conv2_gemm<<<(BN_ * H2_ * H2_) / 128, 256, 288 * 34 * 2, stream>>>(pool1, Wc2, b2, c2raw);
  conv2_stats<<<2048, 256, 0, stream>>>(c2raw, stats2);
  bnpool2<<<NND_ / 256, 256, 0, stream>>>(c2raw, g2, be2, stats2, xA);

  // ---- GAT layer 1: D=32, F=32, HF=64, NC=80 ----
  prep_wcat<<<(64 * 80) / 256, 256, 0, stream>>>(fW[0], fb[0], wW[0], wb[0],
                                                 Wcat, biascat, 64, 32, 64, 80);
  zerof<<<4096, 256, 0, stream>>>(num, NND_ * 64);
  zerof<<<1024, 256, 0, stream>>>(den, NND_ * 2);
  init_amax<<<1, 32, 0, stream>>>(amaxU);
  gat_gemm<2, 5, 64, 32><<<EED_ / 128, 256, 64 * 82 * 2, stream>>>(
      xA, src, tgt, Wcat, biascat, ybuf, abuf, amaxU);
  gat_scatter<<<(EED_ * 2) / 256, 256, 0, stream>>>(ybuf, abuf, amaxU, tgt, num, den, 32, 64);
  gat_norm<<<(NND_ * 64) / 256, 256, 0, stream>>>(num, den, xB, nullptr, 32, 64);

  // ---- GAT layer 2: D=64, F=64, HF=128, NC=144 ----
  prep_wcat<<<(128 * 144) / 256, 256, 0, stream>>>(fW[1], fb[1], wW[1], wb[1],
                                                   Wcat, biascat, 128, 64, 128, 144);
  zerof<<<4096, 256, 0, stream>>>(num, NND_ * 128);
  zerof<<<1024, 256, 0, stream>>>(den, NND_ * 2);
  init_amax<<<1, 32, 0, stream>>>(amaxU);
  gat_gemm<4, 9, 128, 64><<<EED_ / 128, 256, 128 * 146 * 2, stream>>>(
      xB, src, tgt, Wcat, biascat, ybuf, abuf, amaxU);
  gat_scatter<<<(EED_ * 2) / 256, 256, 0, stream>>>(ybuf, abuf, amaxU, tgt, num, den, 64, 128);
  gat_norm<<<(NND_ * 128) / 256, 256, 0, stream>>>(num, den, xA, nullptr, 64, 128);

  // ---- GAT layer 3: D=128, F=64, HF=128, NC=144; fused graph pooling ----
  prep_wcat<<<(256 * 144) / 256, 256, 0, stream>>>(fW[2], fb[2], wW[2], wb[2],
                                                   Wcat, biascat, 256, 64, 128, 144);
  zerof<<<4096, 256, 0, stream>>>(num, NND_ * 128);
  zerof<<<1024, 256, 0, stream>>>(den, NND_ * 2);
  zerof<<<16, 256, 0, stream>>>(gsum, 32 * 128);
  init_amax<<<1, 32, 0, stream>>>(amaxU);
  gat_gemm<8, 9, 128, 128><<<EED_ / 128, 256, 256 * 146 * 2, stream>>>(
      xA, src, tgt, Wcat, biascat, ybuf, abuf, amaxU);
  gat_scatter<<<(EED_ * 2) / 256, 256, 0, stream>>>(ybuf, abuf, amaxU, tgt, num, den, 64, 128);
  gat_norm<<<(NND_ * 128) / 256, 256, 0, stream>>>(num, den, nullptr, gsum, 64, 128);

  // ---- MLP head ----
  mlp_head<<<1, 256, 0, stream>>>(gsum, m1W, m1b, m2W, m2b, out);
}
